// StereoDepthPredictor_6201932775925
// MI455X (gfx1250) — matrix operand order repaired and checked by reading, not compiled, not run on hardware
//
#include <hip/hip_runtime.h>

// ---------------------------------------------------------------------------
// CDNA5 (gfx1250) types for WMMA bf16: A/B = 16 bf16 (8 VGPR), C/D = 8 f32.
// ---------------------------------------------------------------------------
typedef __attribute__((ext_vector_type(16))) __bf16 v16bf;
typedef __attribute__((ext_vector_type(8)))  float  v8f;
typedef __attribute__((ext_vector_type(4)))  unsigned int v4u;
typedef __attribute__((ext_vector_type(8)))  unsigned int v8u;

struct AW32 { v4u lo, hi; };
static_assert(sizeof(AW32) == 32, "AW32 must be 32 bytes");

// ---------------------------------------------------------------------------
// Cost volume (PSM cosine): out[b,d,h,w] = mean_c l[b,c,h,w] * r[b,c,h,w-d]
// LDS-tiled: one (b,h) row, 64-wide w tile, C in chunks of 32; every chunk is
// loaded once and reused by all D disparities (memory-bound stage).
// ---------------------------------------------------------------------------
__global__ __launch_bounds__(256) void k_cost_volume(
    const float* __restrict__ L, const float* __restrict__ R,
    float* __restrict__ out, int C, int D, int H, int W, int Ctot, int chOff)
{
    __shared__ float ls[32][65];   // padded to dodge bank conflicts
    __shared__ float rs[32][89];   // width 64 + (D-1) <= 87

    const int row = blockIdx.x;
    const int b = row / H, h = row % H;
    const int w0 = blockIdx.y * 64;
    const int tid = threadIdx.x;
    const int RW = 64 + D - 1;
    const size_t cs = (size_t)H * W;
    const float* lrow = L + ((size_t)b * C * H + h) * W;
    const float* rrow = R + ((size_t)b * C * H + h) * W;

    float acc[6] = {0.f, 0.f, 0.f, 0.f, 0.f, 0.f};

    for (int c0 = 0; c0 < C; c0 += 32) {
        for (int j = tid; j < 32 * 64; j += 256) {
            int cc = j >> 6, wl = j & 63;
            int wg = w0 + wl;
            ls[cc][wl] = (wg < W) ? lrow[(size_t)(c0 + cc) * cs + wg] : 0.f;
        }
        for (int j = tid; j < 32 * RW; j += 256) {
            int cc = j / RW, wl = j % RW;
            int wg = w0 - (D - 1) + wl;
            rs[cc][wl] = (wg >= 0 && wg < W) ? rrow[(size_t)(c0 + cc) * cs + wg] : 0.f;
        }
        __syncthreads();
        for (int j = 0; j < 6; ++j) {
            int i = tid + j * 256;
            if (i < D * 64) {
                int d = i >> 6, wl = i & 63;
                int rj = wl + (D - 1) - d;
                float a = acc[j];
                #pragma unroll
                for (int cc = 0; cc < 32; ++cc) a += ls[cc][wl] * rs[cc][rj];
                acc[j] = a;
            }
        }
        __syncthreads();
    }

    const float inv = 1.f / (float)C;
    for (int j = 0; j < 6; ++j) {
        int i = tid + j * 256;
        if (i < D * 64) {
            int d = i >> 6, wl = i & 63;
            int wg = w0 + wl;
            if (wg < W)
                out[((size_t)(b * Ctot + chOff + d) * H + h) * W + wg] = acc[j] * inv;
        }
    }
}

// ---------------------------------------------------------------------------
// Convert a channel slice of an NCHW f32 tensor to NHWC bf16, channels padded
// to Cpad (zero fill) so WMMA K-steps of 32 are contiguous 64B rows.
// ---------------------------------------------------------------------------
__global__ void k_nchw2nhwc_bf16(const float* __restrict__ src, __bf16* __restrict__ dst,
                                 int B, int H, int W, int Cin, int Ctot, int chOff, int Cpad)
{
    int i = blockIdx.x * blockDim.x + threadIdx.x;
    int total = B * H * W * Cpad;
    if (i >= total) return;
    int c = i % Cpad, p = i / Cpad;               // p = b*H*W + h*W + w
    float v = 0.f;
    if (c < Cin) {
        int b = p / (H * W), hw = p % (H * W);
        v = src[(size_t)(b * Ctot + chOff + c) * H * W + hw];
    }
    dst[i] = (__bf16)v;
}

// OIHW f32 -> [tap][Opad][Cpad] bf16 (zero padded).
__global__ void k_pack_w(const float* __restrict__ w, __bf16* __restrict__ dst,
                         int O, int I, int taps, int Opad, int Cpad)
{
    int i = blockIdx.x * blockDim.x + threadIdx.x;
    int total = taps * Opad * Cpad;
    if (i >= total) return;
    int ci = i % Cpad, r = i / Cpad, o = r % Opad, t = r / Opad;
    float v = 0.f;
    if (o < O && ci < I) v = w[(size_t)(o * I + ci) * taps + t];
    dst[i] = (__bf16)v;
}

// ---------------------------------------------------------------------------
// Implicit-GEMM conv (3x3 pad 1, or 1x1 pad 0) with v_wmma_f32_16x16x32_bf16.
// One wave computes a 16(Cout) x 64(spatial) tile: 4 accumulators, A fragment
// loaded once per K step and reused for 4 WMMAs (5 loads / 4 wmma).
// A (weights) fragment, contiguous k order (same as B):
//   lanes 0-15: M=lane, halves = K[k..k+7], K[k+8..k+15]
//   lanes 16-31: M=lane-16, halves = K[k+16..k+23], K[k+24..k+31]
// B (activations) fragment: lanes 0-15: N=lane, halves K[k..k+15];
//   lanes 16-31: N=lane-16, halves K[k+16..k+31] (contiguous NHWC load).
// D epilogue (32-bit C/D layout): vgpr v, lane l -> m=v+8*(l>=16), n=l%16.
// Fuses conv-bias, BN scale/shift, residual add, ReLU.
// ---------------------------------------------------------------------------
__global__ __launch_bounds__(256) void k_conv_wmma(
    const __bf16* __restrict__ X, const __bf16* __restrict__ Wp,
    float* __restrict__ out,
    int B, int H, int W, int Cpad, int Cout, int taps, int Opad,
    const float* __restrict__ scale, const float* __restrict__ shift,
    const float* __restrict__ cbias,
    const float* __restrict__ res, int resCtot, int resOff,
    int outCtot, int outOff, int doRelu)
{
    const int lane = threadIdx.x & 31;
    const int wv = threadIdx.x >> 5;
    const int coTile = blockIdx.y * 8 + wv;
    if (coTile * 16 >= Opad) return;            // wave-uniform: EXEC stays all-1s

    const int HW = H * W;
    const int nBase = blockIdx.x * 64;
    const int hi = (lane >= 16) ? 1 : 0;
    const int kSelB = hi ? 16 : 0;
    const int aSel = hi ? 16 : 0;
    const int aRow = coTile * 16 + (lane & 15);
    const size_t tapStride = (size_t)Opad * Cpad;
    const __bf16* wbase = Wp + (size_t)aRow * Cpad;

    // spatial decode for the 4 N sub-tiles (16 positions each)
    int bj[4], hwj[4], hj[4], wj[4];
    #pragma unroll
    for (int j = 0; j < 4; ++j) {
        int n = nBase + j * 16 + (lane & 15);
        bj[j] = n / HW;
        hwj[j] = n % HW;
        hj[j] = hwj[j] / W;
        wj[j] = hwj[j] % W;
    }

    const v8f zf = {0.f, 0.f, 0.f, 0.f, 0.f, 0.f, 0.f, 0.f};
    const v8u zu = {0u, 0u, 0u, 0u, 0u, 0u, 0u, 0u};
    v8f acc[4];
    #pragma unroll
    for (int j = 0; j < 4; ++j) acc[j] = zf;

    for (int t = 0; t < taps; ++t) {
        const int dy = (taps == 9) ? (t / 3 - 1) : 0;
        const int dx = (taps == 9) ? (t % 3 - 1) : 0;
        const __bf16* wrow = wbase + (size_t)t * tapStride;

        const __bf16* xrow[4];
        bool inb[4];
        #pragma unroll
        for (int j = 0; j < 4; ++j) {
            int y = hj[j] + dy, x = wj[j] + dx;
            inb[j] = (y >= 0) && (y < H) && (x >= 0) && (x < W);
            xrow[j] = X + ((size_t)(bj[j] * H + y) * W + x) * Cpad + kSelB;
            if (inb[j]) __builtin_prefetch(xrow[j], 0, 0);   // global_prefetch_b8
        }

        for (int k = 0; k < Cpad; k += 32) {
            AW32 aw;
            aw.lo = *(const v4u*)(wrow + k + aSel);
            aw.hi = *(const v4u*)(wrow + k + aSel + 8);
            v16bf av = __builtin_bit_cast(v16bf, aw);

            #pragma unroll
            for (int j = 0; j < 4; ++j) {
                v8u bu = zu;
                if (inb[j]) bu = *(const v8u*)(xrow[j] + k);
                v16bf bv = __builtin_bit_cast(v16bf, bu);
                acc[j] = __builtin_amdgcn_wmma_f32_16x16x32_bf16(
                    false, av, false, bv, (short)0, acc[j], false, false);
            }
        }
    }

    #pragma unroll
    for (int j = 0; j < 4; ++j) {
        #pragma unroll
        for (int v = 0; v < 8; ++v) {
            int co = coTile * 16 + v + hi * 8;
            if (co >= Cout) continue;
            float val = acc[j][v];
            if (cbias) val += cbias[co];
            if (scale) val = val * scale[co] + shift[co];
            if (res)   val += res[(size_t)(bj[j] * resCtot + resOff + co) * HW + hwj[j]];
            if (doRelu) val = fmaxf(val, 0.f);
            out[(size_t)(bj[j] * outCtot + outOff + co) * HW + hwj[j]] = val;
        }
    }
}

// Depthwise 3x3 conv + BN + ReLU (f32, NCHW with channel offsets).
__global__ void k_dwconv3x3(const float* __restrict__ x, const float* __restrict__ w,
                            const float* __restrict__ scale, const float* __restrict__ shift,
                            float* __restrict__ out, int B, int C, int H, int W,
                            int inCtot, int inOff, int outCtot, int outOff)
{
    int i = blockIdx.x * blockDim.x + threadIdx.x;
    int total = B * C * H * W;
    if (i >= total) return;
    int wc = i % W, h = (i / W) % H, c = (i / (W * H)) % C, b = i / (W * H * C);
    const float* xc = x + (size_t)(b * inCtot + inOff + c) * H * W;
    float s = 0.f;
    #pragma unroll
    for (int kh = 0; kh < 3; ++kh)
        #pragma unroll
        for (int kw = 0; kw < 3; ++kw) {
            int y = h + kh - 1, xx = wc + kw - 1;
            if (y >= 0 && y < H && xx >= 0 && xx < W)
                s += xc[y * W + xx] * w[c * 9 + kh * 3 + kw];
        }
    s = fmaxf(s * scale[c] + shift[c], 0.f);
    out[(size_t)(b * outCtot + outOff + c) * H * W + h * W + wc] = s;
}

// 2x2 average pool (input H,W; output H/2,W/2).
__global__ void k_avgpool2(const float* __restrict__ x, float* __restrict__ out,
                           int B, int C, int H, int W)
{
    int Ho = H / 2, Wo = W / 2;
    int i = blockIdx.x * blockDim.x + threadIdx.x;
    int total = B * C * Ho * Wo;
    if (i >= total) return;
    int wo = i % Wo, ho = (i / Wo) % Ho, c = (i / (Wo * Ho)) % C, b = i / (Wo * Ho * C);
    const float* p = x + (size_t)(b * C + c) * H * W + (2 * ho) * W + 2 * wo;
    out[i] = 0.25f * (p[0] + p[1] + p[W] + p[W + 1]);
}

// 2x bilinear upsample, align-corners (matches reference index math).
__global__ void k_up2(const float* __restrict__ x, float* __restrict__ out,
                      int B, int C, int H, int W)
{
    int H2 = 2 * H, W2 = 2 * W;
    int i = blockIdx.x * blockDim.x + threadIdx.x;
    int total = B * C * H2 * W2;
    if (i >= total) return;
    int ox = i % W2, oy = (i / W2) % H2, c = (i / (W2 * H2)) % C, b = i / (W2 * H2 * C);
    float yc = (float)oy * (float)(H - 1) / (float)(2 * H - 1);
    float xc = (float)ox * (float)(W - 1) / (float)(2 * W - 1);
    int y0 = (int)yc, x0 = (int)xc;
    int y1 = (y0 + 1 < H) ? y0 + 1 : H - 1;
    int x1 = (x0 + 1 < W) ? x0 + 1 : W - 1;
    float wy = yc - (float)y0, wx = xc - (float)x0;
    const float* p = x + (size_t)(b * C + c) * H * W;
    float top = p[y0 * W + x0] * (1.f - wx) + p[y0 * W + x1] * wx;
    float bot = p[y1 * W + x0] * (1.f - wx) + p[y1 * W + x1] * wx;
    out[i] = top * (1.f - wy) + bot * wy;
}

// Copy C channels between NCHW tensors with differing channel totals/offsets.
__global__ void k_copy_ch(const float* __restrict__ src, float* __restrict__ dst,
                          int B, int C, int H, int W,
                          int srcCtot, int srcOff, int dstCtot, int dstOff)
{
    int i = blockIdx.x * blockDim.x + threadIdx.x;
    int total = B * C * H * W;
    if (i >= total) return;
    int hw = i % (H * W), c = (i / (H * W)) % C, b = i / (H * W * C);
    dst[(size_t)(b * dstCtot + dstOff + c) * H * W + hw] =
        src[(size_t)(b * srcCtot + srcOff + c) * H * W + hw];
}

// ---------------------------------------------------------------------------
// Host orchestration. Input order = setup_inputs() recursive insertion order:
//  0..3  left_8, right_8, left_16, right_16
//  4..9  e2s_rg: pcw, pcbn(s,b), cow, cobn(s,b)
// 10..15 e2s_bb: c1w, bn1(s,b), c2w, bn2(s,b)
// 16..21 dr_rg:  pcw, pcbn(s,b), cow, cobn(s,b)
// 22..27 dr_bb:  c1w, bn1(s,b), c2w, bn2(s,b)
// 28..37 do_w1, do_b1, do_bn1(s,b), do_w2, do_b2, do_bn2(s,b), do_w3, do_b3
// ---------------------------------------------------------------------------
extern "C" void kernel_launch(void* const* d_in, const int* in_sizes, int n_in,
                              void* d_out, int out_size, void* d_ws, size_t ws_size,
                              hipStream_t stream)
{
    (void)in_sizes; (void)n_in; (void)out_size; (void)ws_size;
    const int B = 8;
    const float* left8   = (const float*)d_in[0];
    const float* right8  = (const float*)d_in[1];
    const float* left16  = (const float*)d_in[2];
    const float* right16 = (const float*)d_in[3];
    auto P_ = [&](int i) { return (const float*)d_in[i]; };

    char* ws = (char*)d_ws;
    constexpr size_t OFF_G1 = 0;                        // (8,72,48,160)  f32
    constexpr size_t OFF_P  = OFF_G1 + 17694720;        // (8,72,24,80)
    constexpr size_t OFF_T  = OFF_P  + 4423680;         // (8,72,24,80)
    constexpr size_t OFF_G2 = OFF_T  + 4423680;         // (8,84,24,80)
    constexpr size_t OFF_G3 = OFF_G2 + 5160960;         // (8,252,24,80)
    constexpr size_t OFF_T2 = OFF_G3 + 15482880;        // (8,252,24,80)
    constexpr size_t OFF_U1 = OFF_T2 + 15482880;        // (8,252,48,160)
    constexpr size_t OFF_D1 = OFF_U1 + 61931520;        // (8,126,48,160)
    constexpr size_t OFF_U2 = OFF_D1 + 30965760;        // (8,126,96,320)
    constexpr size_t OFF_D2 = OFF_U2 + 123863040;       // (8,63,96,320)
    constexpr size_t OFF_XB = OFF_D2 + 61931520;        // NHWC bf16 scratch (64MB)
    constexpr size_t OFF_WP = OFF_XB + 67108864;        // packed weights (2MB)

    float* G1 = (float*)(ws + OFF_G1);
    float* Pb = (float*)(ws + OFF_P);
    float* Tb = (float*)(ws + OFF_T);
    float* G2 = (float*)(ws + OFF_G2);
    float* G3 = (float*)(ws + OFF_G3);
    float* T2 = (float*)(ws + OFF_T2);
    float* U1 = (float*)(ws + OFF_U1);
    float* D1 = (float*)(ws + OFF_D1);
    float* U2 = (float*)(ws + OFF_U2);
    float* D2 = (float*)(ws + OFF_D2);
    __bf16* XB = (__bf16*)(ws + OFF_XB);
    __bf16* WP = (__bf16*)(ws + OFF_WP);
    float* psv   = (float*)d_out;                       // (8,252,24,80)
    float* depth = (float*)d_out + 3870720;             // (8,96,96,320)

    auto cdiv = [](int a, int b) { return (a + b - 1) / b; };
    auto convert = [&](const float* src, int H, int W, int Cin, int Ctot, int chOff, int Cpad) {
        int total = B * H * W * Cpad;
        k_nchw2nhwc_bf16<<<cdiv(total, 256), 256, 0, stream>>>(src, XB, B, H, W, Cin, Ctot, chOff, Cpad);
    };
    auto pack = [&](const float* w, int O, int I, int taps, int Opad, int Cpad) {
        int total = taps * Opad * Cpad;
        k_pack_w<<<cdiv(total, 256), 256, 0, stream>>>(w, WP, O, I, taps, Opad, Cpad);
    };
    auto conv = [&](float* out, int H, int W, int Cpad, int Cout, int taps, int Opad,
                    const float* sc, const float* sh, const float* cb,
                    const float* res, int resCtot, int resOff,
                    int outCtot, int outOff, int relu) {
        dim3 grid((B * H * W) / 64, cdiv(Opad / 16, 8));
        k_conv_wmma<<<grid, 256, 0, stream>>>(XB, WP, out, B, H, W, Cpad, Cout, taps, Opad,
                                              sc, sh, cb, res, resCtot, resOff, outCtot, outOff, relu);
    };

    // 1) cost volumes -> concat slots
    k_cost_volume<<<dim3(B * 48, cdiv(160, 64)), 256, 0, stream>>>(left8, right8, G1, 256, 24, 48, 160, 72, 0);
    k_cost_volume<<<dim3(B * 24, cdiv(80, 64)), 256, 0, stream>>>(left16, right16, G2, 256, 12, 24, 80, 84, 72);

    // 2) e2s res-ghost: x1 = relu(bn(conv3x3)) -> G1[24:48]; x2 = relu(bn(dw3x3)) -> G1[48:72]
    pack(P_(4), 24, 24, 9, 32, 32);
    convert(G1, 48, 160, 24, 72, 0, 32);
    conv(G1, 48, 160, 32, 24, 9, 32, P_(5), P_(6), nullptr, nullptr, 0, 0, 72, 24, 1);
    k_dwconv3x3<<<cdiv(B * 24 * 48 * 160, 256), 256, 0, stream>>>(G1, P_(7), P_(8), P_(9), G1,
                                                                  B, 24, 48, 160, 72, 24, 72, 48);

    // 3) avgpool2 -> P (8,72,24,80)
    k_avgpool2<<<cdiv(B * 72 * 24 * 80, 256), 256, 0, stream>>>(G1, Pb, B, 72, 48, 160);

    // 4) e2s basic block -> G2[0:72] (residual add + relu fused)
    pack(P_(10), 72, 72, 9, 80, 96);
    convert(Pb, 24, 80, 72, 72, 0, 96);
    conv(Tb, 24, 80, 96, 72, 9, 80, P_(11), P_(12), nullptr, nullptr, 0, 0, 72, 0, 1);
    pack(P_(13), 72, 72, 9, 80, 96);
    convert(Tb, 24, 80, 72, 72, 0, 96);
    conv(G2, 24, 80, 96, 72, 9, 80, P_(14), P_(15), nullptr, Pb, 72, 0, 84, 0, 1);

    // 5) concat: G2(84) -> G3[0:84]
    k_copy_ch<<<cdiv(B * 84 * 24 * 80, 256), 256, 0, stream>>>(G2, G3, B, 84, 24, 80, 84, 0, 252, 0);

    // 6) dr res-ghost: x1 -> G3[84:168]; x2 -> G3[168:252]
    pack(P_(16), 84, 84, 9, 96, 96);
    convert(G2, 24, 80, 84, 84, 0, 96);
    conv(G3, 24, 80, 96, 84, 9, 96, P_(17), P_(18), nullptr, nullptr, 0, 0, 252, 84, 1);
    k_dwconv3x3<<<cdiv(B * 84 * 24 * 80, 256), 256, 0, stream>>>(G3, P_(19), P_(20), P_(21), G3,
                                                                 B, 84, 24, 80, 252, 84, 252, 168);

    // 7) dr basic block -> psv_16 (output 0)
    pack(P_(22), 252, 252, 9, 256, 256);
    convert(G3, 24, 80, 252, 252, 0, 256);
    conv(T2, 24, 80, 256, 252, 9, 256, P_(23), P_(24), nullptr, nullptr, 0, 0, 252, 0, 1);
    pack(P_(25), 252, 252, 9, 256, 256);
    convert(T2, 24, 80, 252, 252, 0, 256);
    conv(psv, 24, 80, 256, 252, 9, 256, P_(26), P_(27), nullptr, G3, 252, 0, 252, 0, 1);

    // 8) decoder: up2 -> conv(252->126)+b+bn+relu -> up2 -> conv(126->63)+b+bn+relu -> 1x1(63->96)+b
    k_up2<<<cdiv(B * 252 * 48 * 160, 256), 256, 0, stream>>>(psv, U1, B, 252, 24, 80);
    pack(P_(28), 126, 252, 9, 128, 256);
    convert(U1, 48, 160, 252, 252, 0, 256);
    conv(D1, 48, 160, 256, 126, 9, 128, P_(30), P_(31), P_(29), nullptr, 0, 0, 126, 0, 1);

    k_up2<<<cdiv(B * 126 * 96 * 320, 256), 256, 0, stream>>>(D1, U2, B, 126, 48, 160);
    pack(P_(32), 63, 126, 9, 64, 128);
    convert(U2, 96, 320, 126, 126, 0, 128);
    conv(D2, 96, 320, 128, 63, 9, 64, P_(34), P_(35), P_(33), nullptr, 0, 0, 63, 0, 1);

    pack(P_(36), 96, 63, 1, 96, 64);
    convert(D2, 96, 320, 63, 63, 0, 64);
    conv(depth, 96, 320, 64, 96, 1, 96, nullptr, nullptr, P_(37), nullptr, 0, 0, 96, 0, 0);
}
